// CRNN_2413771620417
// MI455X (gfx1250) — compile-verified, matrix operand order checked by reading, not run on hardware
//
#include <hip/hip_runtime.h>

typedef __attribute__((ext_vector_type(2)))  float  v2f;
typedef __attribute__((ext_vector_type(8)))  float  v8f;
typedef __attribute__((ext_vector_type(16))) __bf16 v16bf;

namespace {
constexpr int   kNS = 32;
constexpr int   kNR = 256;
constexpr int   kNG = 2048;
constexpr float kR  = 0.0019872036f;
}

// w_v element for column k (k in [0,36): 0..31 = log(clip(u)), 32 = -1/(RT), 33 = log T, 34/35 = pad)
__device__ __forceinline__ float wv_elem(int k, const float* __restrict__ urow,
                                         float negInvRT, float logT) {
  if (k < kNS)      return __logf(fminf(fmaxf(urow[k], 1e-5f), 60.0f));
  if (k == kNS)     return negInvRT;
  if (k == kNS + 1) return logT;
  return 0.0f;
}

__global__ __launch_bounds__(256, 1) void crnn_wmma_kernel(
    const float* __restrict__ u, const float* __restrict__ t,
    const float* __restrict__ tsteps, const float* __restrict__ T_tab,
    const float* __restrict__ w_in, const float* __restrict__ w_b,
    const float* __restrict__ w_out, float* __restrict__ out, int nB)
{
  // Weights staged once per block, pre-swizzled into WMMA fragment order.
  __shared__ float  s_win[9 * 16 * 32 * 2];    // 36 KB: GEMM1 B frags, f32, K padded 34->36
  __shared__ __bf16 s_wout[8 * 2 * 32 * 16];   // 16 KB: GEMM2 B frags, bf16
  __shared__ float  s_wb[kNR];                 //  1 KB: bias
  __shared__ __bf16 s_stage[8 * 16 * 32];      //  8 KB: per-wave exp() transpose tiles

  const int tid  = threadIdx.x;
  const int wave = tid >> 5;
  const int lane = tid & 31;
  const int lmod = lane & 15;
  const int g    = lane >> 4;

  // ---- stage w_in in f32 16x16x4 B-fragment order: lane holds N=l%16, K=2*(l/16)+v ----
  for (int i = tid; i < 9 * 16 * 32 * 2; i += 256) {
    int v  = i & 1;
    int l  = (i >> 1) & 31;
    int nt = (i >> 6) & 15;
    int ks = i >> 10;
    int k  = ks * 4 + (l >> 4) * 2 + v;
    int r  = nt * 16 + (l & 15);
    s_win[i] = (k < kNS + 2) ? w_in[k * kNR + r] : 0.0f;
  }
  // ---- stage w_out^T in bf16 16x16x32 B-fragment order ----
  for (int i = tid; i < 8 * 2 * 32 * 16; i += 256) {
    int j  = i & 15;
    int l  = (i >> 4) & 31;
    int nt = (i >> 9) & 1;
    int p  = i >> 10;
    int v  = j >> 1, jj = j & 1, lg = l >> 4;
    int kb = (v < 4) ? (lg * 8 + 2 * v) : (16 + lg * 8 + 2 * (v - 4));
    int k  = p * 32 + kb + jj;
    int n  = nt * 16 + (l & 15);
    s_wout[i] = (__bf16)w_out[n * kNR + k];
  }
  if (tid < kNR) s_wb[tid] = w_b[tid];
  __syncthreads();

  const int base0 = blockIdx.x * 128 + wave * 16;
  int m = base0 + lmod;
  if (m >= nB) m = nB - 1;

  // T(t): linear interpolation on the sorted uniform time grid (jnp.interp semantics)
  float tm  = t[m];
  float pos = tm * (float)(kNG - 1);
  pos = fminf(fmaxf(pos, 0.0f), (float)(kNG - 1));
  int i0 = (int)pos;
  if (i0 > kNG - 2) i0 = kNG - 2;
  float ta = tsteps[i0], tb = tsteps[i0 + 1];
  float fr = fminf(fmaxf((tm - ta) / (tb - ta), 0.0f), 1.0f);
  float T  = T_tab[i0] + fr * (T_tab[i0 + 1] - T_tab[i0]);
  float negInvRT = -1.0f / (kR * T);
  float logT     = __logf(T);

  // Nine 16x4 f32 A fragments of w_v: lane holds M=l%16, K=2*(l/16)+v
  const float* urow = &u[(size_t)m * kNS];
  v2f afr[9];
  #pragma unroll
  for (int ks = 0; ks < 9; ++ks) {
    int k0 = ks * 4 + g * 2;
    afr[ks].x = wv_elem(k0,     urow, negInvRT, logT);
    afr[ks].y = wv_elem(k0 + 1, urow, negInvRT, logT);
  }

  __bf16* stage = &s_stage[wave * (16 * 32)];
  v8f dacc0 = {};
  v8f dacc1 = {};

  for (int p = 0; p < 8; ++p) {
    // GEMM1: two 16-wide reaction tiles (one 32-wide K-chunk of GEMM2)
    #pragma unroll
    for (int hf = 0; hf < 2; ++hf) {
      const int nt = p * 2 + hf;
      v8f acc = {};
      #pragma unroll
      for (int ks = 0; ks < 9; ++ks) {
        const float* bp = &s_win[((ks * 16 + nt) * 32 + lane) * 2];
        v2f b;
        b.x = bp[0];
        b.y = bp[1];
        acc = __builtin_amdgcn_wmma_f32_16x16x4_f32(
            false, afr[ks], false, b, (short)0, acc, false, false);
      }
      // epilogue: + bias, clamp to [-30,30], exp, bf16, stage for transpose
      float bias = s_wb[nt * 16 + lmod];
      #pragma unroll
      for (int v = 0; v < 8; ++v) {
        float x = acc[v] + bias;
        x = fminf(fmaxf(x, -30.0f), 30.0f);
        stage[(v + 8 * g) * 32 + hf * 16 + lmod] = (__bf16)__expf(x);
      }
    }

    // cross-lane same-wave RAW through LDS: enforce DS completion + compiler order
    asm volatile("s_wait_dscnt 0" ::: "memory");

    // Build bf16 16x32 A fragment from staged tile: M=l%16, K per ISA layout
    v16bf a2;
    {
      const __bf16* sp = &stage[lmod * 32];
      #pragma unroll
      for (int v = 0; v < 8; ++v) {
        int kb = (v < 4) ? (g * 8 + 2 * v) : (16 + g * 8 + 2 * (v - 4));
        a2[2 * v]     = sp[kb];
        a2[2 * v + 1] = sp[kb + 1];
      }
    }
    // GEMM2: accumulate this K-chunk into both 16-wide species tiles
    {
      const __bf16* wp0 = &s_wout[((p * 2 + 0) * 32 + lane) * 16];
      const __bf16* wp1 = &s_wout[((p * 2 + 1) * 32 + lane) * 16];
      v16bf b0, b1;
      #pragma unroll
      for (int j = 0; j < 16; ++j) { b0[j] = wp0[j]; b1[j] = wp1[j]; }
      dacc0 = __builtin_amdgcn_wmma_f32_16x16x32_bf16(
          false, a2, false, b0, (short)0, dacc0, false, false);
      dacc1 = __builtin_amdgcn_wmma_f32_16x16x32_bf16(
          false, a2, false, b1, (short)0, dacc1, false, false);
    }
  }

  // clamp du to [-1e4, 1e4] and store (C/D layout: M = v + 8*(lane/16), N = lane%16)
  #pragma unroll
  for (int v = 0; v < 8; ++v) {
    int mrow = base0 + v + 8 * g;
    if (mrow < nB) {
      out[(size_t)mrow * kNS + lmod]      = fminf(fmaxf(dacc0[v], -1e4f), 1e4f);
      out[(size_t)mrow * kNS + 16 + lmod] = fminf(fmaxf(dacc1[v], -1e4f), 1e4f);
    }
  }
}

extern "C" void kernel_launch(void* const* d_in, const int* in_sizes, int n_in,
                              void* d_out, int out_size, void* d_ws, size_t ws_size,
                              hipStream_t stream) {
  (void)n_in; (void)out_size; (void)d_ws; (void)ws_size;
  const float* u      = (const float*)d_in[0];
  const float* t      = (const float*)d_in[1];
  const float* tsteps = (const float*)d_in[2];
  const float* T_tab  = (const float*)d_in[3];
  /* d_in[4] = P_tab: computed-but-unused in the reference */
  const float* w_in   = (const float*)d_in[5];
  const float* w_b    = (const float*)d_in[6];
  const float* w_out  = (const float*)d_in[7];
  float* out = (float*)d_out;

  int nB = in_sizes[1];                 // t has one entry per sample
  int blocks = (nB + 127) / 128;        // 128 samples per 256-thread block (8 wave32)
  crnn_wmma_kernel<<<blocks, 256, 0, stream>>>(u, t, tsteps, T_tab,
                                               w_in, w_b, w_out, out, nB);
}
